// PQLayer_66142496358463
// MI455X (gfx1250) — matrix operand order repaired and checked by reading, not compiled, not run on hardware
//
#include <hip/hip_runtime.h>

typedef __attribute__((ext_vector_type(16))) _Float16 v16h;
typedef __attribute__((ext_vector_type(8)))  float    v8f;

union Frag16 { v16h h; unsigned int u[8]; };

#define NB    65536
#define FEAT  512
#define MG    8
#define KC    256
#define DD    64
#define WAVES 8
#define RPB   128        // rows per block (16 per wave)

// LDS strides (in element units) chosen for conflict-free 16-lane gathers
#define CRM_S 70         // halves: C row-major  [256][70]
#define CT_S  262        // halves: C transposed [64][262]
#define XF_S  66         // floats: x f32 tile   [16][66] per wave
#define CW_S  268        // halves: codes f16    [16][268] per wave
#define WREG_B 8576      // per-wave scratch bytes = max(16*66*4, 16*268*2)

__global__ __launch_bounds__(256)
void pq_soft_assign_kernel(const float* __restrict__ x,
                           const float* __restrict__ C,
                           float* __restrict__ out)
{
    __shared__ _Float16 sCrm[KC * CRM_S];                  // 35840 B
    __shared__ _Float16 sCt [DD * CT_S];                   // 33536 B
    __shared__ unsigned long long sWaveRaw[WAVES * (WREG_B / 8)]; // 68608 B

    const int tid  = threadIdx.x;
    const int wave = tid >> 5;
    const int lane = tid & 31;
    const int m    = blockIdx.y;
    const int hh   = lane >> 4;        // hi half of wave
    const int ln15 = lane & 15;
    const int kb   = hh ? 8 : 0;       // A-fragment K offset for hi lanes

    // ---- Stage C[m] into LDS as f16: row-major and transposed ----
    const float* Cm = C + (size_t)m * KC * DD;
    for (int i = tid; i < KC * DD; i += 256) {
        int k = i >> 6, d = i & 63;
        _Float16 h = (_Float16)Cm[i];
        sCrm[k * CRM_S + d] = h;
        sCt [d * CT_S  + k] = h;
    }

    // ---- Coalesced load of this wave's 16 rows of x into LDS (f32) ----
    float* xf = (float*)&sWaveRaw[(size_t)wave * (WREG_B / 8)];
    const size_t rowG = (size_t)blockIdx.x * RPB + (size_t)wave * 16;
    const float* xBase = x + rowG * FEAT + (size_t)m * DD;
    #pragma unroll
    for (int r = 0; r < 16; ++r) {
        float2 v = *(const float2*)(xBase + (size_t)r * FEAT + lane * 2);
        *(float2*)(xf + r * XF_S + lane * 2) = v;
    }

    __syncthreads();

    // ---- Per-row L2 norm: lane handles row (lane&15), half hh ----
    float inv;
    {
        const float* xr = xf + ln15 * XF_S + hh * 32;
        float s = 0.f;
        #pragma unroll
        for (int i = 0; i < 32; ++i) { float v = xr[i]; s += v * v; }
        s += __shfl_xor(s, 16, 32);
        inv = 1.0f / fmaxf(sqrtf(s), 1e-12f);
    }

    // ---- Build GEMM1 A-fragments (normalized x -> f16) ----
    // A 16x32 layout: lane=row; dword j -> K pair {b,b+1}, b=(j<4?2j:2j+8)+kb
    Frag16 a[2];
    #pragma unroll
    for (int f = 0; f < 2; ++f) {
        #pragma unroll
        for (int j = 0; j < 8; ++j) {
            int base = (j < 4 ? 2 * j : 2 * j + 8) + kb;
            float2 v = *(const float2*)(xf + ln15 * XF_S + f * 32 + base);
            union { unsigned int u; _Float16 h[2]; } pk;
            pk.h[0] = (_Float16)(v.x * inv);
            pk.h[1] = (_Float16)(v.y * inv);
            a[f].u[j] = pk.u;
        }
    }

    // ---- GEMM1: ips[16 rows][256 codewords] ----
    v8f acc[16];
    #pragma unroll
    for (int t = 0; t < 16; ++t) {
        const _Float16* crow = sCrm + (t * 16 + ln15) * CRM_S;
        Frag16 b0, b1;
        #pragma unroll
        for (int j = 0; j < 8; ++j) {
            int bo = (j < 4 ? 2 * j : 2 * j + 8) + (hh ? 8 : 0);   // B layout: lane=col, K pairs in same table order as A
            b0.u[j] = *(const unsigned int*)(crow + bo);
            b1.u[j] = *(const unsigned int*)(crow + 32 + bo);
        }
        v8f c = {0.f, 0.f, 0.f, 0.f, 0.f, 0.f, 0.f, 0.f};
        c = __builtin_amdgcn_wmma_f32_16x16x32_f16(false, a[0].h, false, b0.h, (short)0, c, false, false);
        c = __builtin_amdgcn_wmma_f32_16x16x32_f16(false, a[1].h, false, b1.h, (short)0, c, false, false);
        acc[t] = c;
    }

    // ---- Softmax over k (ALPHA = 1), directly on C/D layout ----
    // acc[t][j] holds row (j + 8*hh), col (16t + ln15)
    #pragma unroll
    for (int j = 0; j < 8; ++j) {
        float mx = acc[0][j];
        #pragma unroll
        for (int t = 1; t < 16; ++t) mx = fmaxf(mx, acc[t][j]);
        #pragma unroll
        for (int off = 8; off > 0; off >>= 1) mx = fmaxf(mx, __shfl_xor(mx, off, 32));
        float s = 0.f;
        #pragma unroll
        for (int t = 0; t < 16; ++t) { float e = __expf(acc[t][j] - mx); acc[t][j] = e; s += e; }
        #pragma unroll
        for (int off = 8; off > 0; off >>= 1) s += __shfl_xor(s, off, 32);
        float is = 1.0f / s;
        #pragma unroll
        for (int t = 0; t < 16; ++t) acc[t][j] *= is;
    }

    // ---- Stream codes to global (f32, non-temporal) + stash f16 in LDS ----
    float* codesOut = out + (size_t)NB * FEAT;
    _Float16* cw = (_Float16*)xf;  // x staging is dead; reuse per-wave region
    #pragma unroll
    for (int j = 0; j < 8; ++j) {
        int rr = j + hh * 8;
        float* orow = codesOut + ((rowG + rr) * MG + m) * KC;
        #pragma unroll
        for (int t = 0; t < 16; ++t) {
            float v = acc[t][j];
            __builtin_nontemporal_store(v, orow + t * 16 + ln15);
            cw[rr * CW_S + t * 16 + ln15] = (_Float16)v;
        }
    }

    // ---- Build GEMM2 A-fragments from codes (f16) ----
    Frag16 ca[8];
    #pragma unroll
    for (int f = 0; f < 8; ++f) {
        #pragma unroll
        for (int j = 0; j < 8; ++j) {
            int base = (j < 4 ? 2 * j : 2 * j + 8) + kb;
            ca[f].u[j] = *(const unsigned int*)(cw + ln15 * CW_S + f * 32 + base);
        }
    }

    // ---- GEMM2: x_hat[16][64] = codes * C[m] ----
    #pragma unroll
    for (int td = 0; td < 4; ++td) {
        const _Float16* ctc = sCt + (td * 16 + ln15) * CT_S;
        v8f c = {0.f, 0.f, 0.f, 0.f, 0.f, 0.f, 0.f, 0.f};
        #pragma unroll
        for (int f = 0; f < 8; ++f) {
            Frag16 bf;
            #pragma unroll
            for (int j = 0; j < 8; ++j)
                bf.u[j] = *(const unsigned int*)(ctc + f * 32 + (j < 4 ? 2 * j : 2 * j + 8) + (hh ? 8 : 0));
            c = __builtin_amdgcn_wmma_f32_16x16x32_f16(false, ca[f].h, false, bf.h, (short)0, c, false, false);
        }
        #pragma unroll
        for (int j = 0; j < 8; ++j) {
            int rr = j + hh * 8;
            __builtin_nontemporal_store(
                c[j], out + (rowG + rr) * FEAT + (size_t)m * DD + td * 16 + ln15);
        }
    }
}

extern "C" void kernel_launch(void* const* d_in, const int* in_sizes, int n_in,
                              void* d_out, int out_size, void* d_ws, size_t ws_size,
                              hipStream_t stream)
{
    (void)in_sizes; (void)n_in; (void)out_size; (void)d_ws; (void)ws_size;
    const float* x = (const float*)d_in[0];   // [65536, 512] f32
    const float* C = (const float*)d_in[1];   // [8, 256, 64] f32
    float* out = (float*)d_out;               // x_hat [65536,512] then codes [65536,8,256]

    dim3 grid(NB / RPB, MG);                  // 512 x 8 blocks
    pq_soft_assign_kernel<<<grid, 256, 0, stream>>>(x, C, out);
}
